// WindowAttention_69672959476064
// MI455X (gfx1250) — matrix operand order repaired and checked by reading, not compiled, not run on hardware
//
#include <hip/hip_runtime.h>
#include <hip/hip_bf16.h>

#define DIM   192
#define NH    6
#define HD    32
#define NTOK  49
#define NPAD  64
#define XSTR  200   // xA/O row stride in halves (400 B -> conflict-free-ish)
#define SSTR  68    // S row stride in floats (272 B -> bank delta 4)
#define PSTR  136   // P row stride in halves (same 272 B rows as S)

typedef _Float16 h4  __attribute__((ext_vector_type(4)));
typedef _Float16 h8  __attribute__((ext_vector_type(8)));
typedef _Float16 h16 __attribute__((ext_vector_type(16)));
typedef float    f8  __attribute__((ext_vector_type(8)));

__device__ __forceinline__ f8 wmma16(h16 a, h16 b, f8 c) {
  // D = A(16x32 f16) * B(32x16 f16) + C(16x16 f32)
  return __builtin_amdgcn_wmma_f32_16x16x32_f16(false, a, false, b, (short)0, c,
                                                false, false);
}

// A operand (16x32 f16): lane holds row m0+(lane&15); K elements
// {kb..kb+7, kb+16..kb+23} with kb = k0 + (lane>=16 ? 8 : 0).
__device__ __forceinline__ h16 ldA(const _Float16* base, int stride, int m0, int k0) {
  int lane = threadIdx.x & 31;
  int row  = m0 + (lane & 15);
  int kb   = k0 + ((lane & 16) ? 8 : 0);
  const _Float16* p = base + row * stride + kb;
  h8 lo = *(const h8*)(p);
  h8 hi = *(const h8*)(p + 16);
  return __builtin_shufflevector(lo, hi, 0,1,2,3,4,5,6,7,8,9,10,11,12,13,14,15);
}

// B operand (32x16 f16): lane holds column n0+(lane&15); K elements
// {kb..kb+7, kb+16..kb+23} with kb = k0 + (lane>=16 ? 8 : 0). Source column-major (K contiguous).
__device__ __forceinline__ h16 ldB(const _Float16* base, int stride, int n0, int k0) {
  int lane = threadIdx.x & 31;
  int col  = n0 + (lane & 15);
  int kb   = k0 + ((lane & 16) ? 8 : 0);
  const _Float16* p = base + col * stride + kb;
  h8 lo = *(const h8*)(p);
  h8 hi = *(const h8*)(p + 16);
  return __builtin_shufflevector(lo, hi, 0,1,2,3,4,5,6,7,8,9,10,11,12,13,14,15);
}

// One-shot: transpose + f32->f16 convert weights into workspace.
__global__ void convert_weights(const float* __restrict__ qkv_w,
                                const float* __restrict__ proj_w,
                                _Float16* __restrict__ wT,
                                _Float16* __restrict__ pT) {
  int i = blockIdx.x * 256 + threadIdx.x;
  if (i < 576 * 192) {
    int c = i / 192, d = i - c * 192;
    wT[i] = (_Float16)qkv_w[d * 576 + c];
  }
  if (i < 192 * 192) {
    int c = i / 192, d = i - c * 192;
    pT[i] = (_Float16)proj_w[d * 192 + c];
  }
}

// Fused window attention: one block per window, 256 threads (8 wave32).
__global__ __launch_bounds__(256) void win_attn_kernel(
    const float* __restrict__ x, const float* __restrict__ qkv_b,
    const float* __restrict__ proj_b, const float* __restrict__ bias_table,
    const int* __restrict__ rel_idx, const _Float16* __restrict__ wT,
    const _Float16* __restrict__ pT, float* __restrict__ out) {
  // LDS layout (157,696 B -> 2 workgroups/WGP):
  //   [0      , 25600) xA : x as f16 64 x (192 pad 200); reused as O later
  //   [25600  , 50176) Qs : 6 x 64x32 f16 (pre-scaled by HD^-0.5)
  //   [50176  , 74752) Ks : 6 x 64x32 f16
  //   [74752  ,105472) Vt : 6 x 32x80 f16 (V transposed, padded stride)
  //   [105472 ,157696) Ss : 3 x 64 x (64 pad 68) f32 scores; P (f16, stride
  //                         136) overwrites each S row in place. Reused at the
  //                         end as compact 64x192 f32 output staging.
  __shared__ __align__(16) unsigned char smem[157696];
  _Float16* xA = (_Float16*)smem;
  _Float16* Qs = (_Float16*)(smem + 25600);
  _Float16* Ks = (_Float16*)(smem + 50176);
  _Float16* Vt = (_Float16*)(smem + 74752);
  float*    Ss = (float*)(smem + 105472);
  _Float16* Ps = (_Float16*)(smem + 105472);

  const int tid  = threadIdx.x;
  const int lane = tid & 31;
  const int wave = tid >> 5;
  const int b    = blockIdx.x;
  const int colL = lane & 15;
  const int rb   = (lane & 16) ? 8 : 0;
  const float SCALE = 0.17677669529663687f;  // 32^-0.5

  // ---- Phase 0: stage x (49x192) -> f16 LDS (float4 loads), zero-pad ----
  {
    const float4* xb4 = (const float4*)(x + (size_t)b * (NTOK * DIM));
    for (int i = tid; i < NPAD * 48; i += 256) {   // 48 float4 per row
      int row = i / 48, c4 = i - row * 48;
      float4 v = {0.f, 0.f, 0.f, 0.f};
      if (row < NTOK) v = xb4[row * 48 + c4];
      h4 hv = {(_Float16)v.x, (_Float16)v.y, (_Float16)v.z, (_Float16)v.w};
      *(h4*)(xA + row * XSTR + c4 * 4) = hv;
    }
  }
  __syncthreads();

  // ---- Phase 1: QKV = x @ qkv_w + b (64x576) -> Q/K/Vt ----
  // Paired M-tiles share the B fragment; two independent WMMA chains.
  for (int p = wave; p < 72; p += 8) {             // 72 pairs = 144 tiles
    int tm0 = (2 * p) & 3;                         // 0 or 2
    int tn  = p >> 1;                              // 0..35 (wave-uniform)
    f8 acc0 = {}, acc1 = {};
    for (int kk = 0; kk < DIM; kk += 32) {
      h16 w  = ldB(wT, DIM, tn * 16, kk);
      h16 a0 = ldA(xA, XSTR, tm0 * 16, kk);
      h16 a1 = ldA(xA, XSTR, tm0 * 16 + 16, kk);
      acc0 = wmma16(a0, w, acc0);
      acc1 = wmma16(a1, w, acc1);
    }
    int col = tn * 16 + colL;
    float bias = qkv_b[col];
    int tok0 = tm0 * 16 + rb, tok1 = tok0 + 16;
    int tnu = __builtin_amdgcn_readfirstlane(tn);  // scalar branch
    if (tnu < 12) {                                // Q
      int hh = col >> 5, d = col & 31;
      _Float16* dq = Qs + hh * 2048 + d;
      for (int r = 0; r < 8; ++r) {
        dq[(tok0 + r) * 32] = (_Float16)((acc0[r] + bias) * SCALE);
        dq[(tok1 + r) * 32] = (_Float16)((acc1[r] + bias) * SCALE);
      }
    } else if (tnu < 24) {                         // K
      int c2 = col - 192; int hh = c2 >> 5, d = c2 & 31;
      _Float16* dk = Ks + hh * 2048 + d;
      for (int r = 0; r < 8; ++r) {
        dk[(tok0 + r) * 32] = (_Float16)(acc0[r] + bias);
        dk[(tok1 + r) * 32] = (_Float16)(acc1[r] + bias);
      }
    } else {                                       // V, transposed [d][tok]
      int c2 = col - 384; int hh = c2 >> 5, d = c2 & 31;
      _Float16* dv = Vt + hh * 2560 + d * 80;
      for (int r = 0; r < 8; ++r) {
        dv[tok0 + r] = (_Float16)(acc0[r] + bias);
        dv[tok1 + r] = (_Float16)(acc1[r] + bias);
      }
    }
  }
  __syncthreads();

  // ---- Phases 2-4, two groups of 3 heads ----
  for (int g = 0; g < 2; ++g) {
    // S = Q @ K^T (per head 64x64; K=32 -> one WMMA per tile, paired on M)
    for (int p = wave; p < 24; p += 8) {           // 24 pairs = 48 tiles
      int t0  = 2 * p;
      int hl  = t0 >> 4;
      int rem = t0 & 15;
      int tm0 = rem & 3, tn = rem >> 2;
      int hh  = g * 3 + hl;
      f8 acc0 = {}, acc1 = {};
      h16 kb = ldB(Ks + hh * 2048, HD, tn * 16, 0);
      h16 a0 = ldA(Qs + hh * 2048, HD, tm0 * 16, 0);
      h16 a1 = ldA(Qs + hh * 2048, HD, tm0 * 16 + 16, 0);
      acc0 = wmma16(a0, kb, acc0);
      acc1 = wmma16(a1, kb, acc1);
      int j = tn * 16 + colL;
      float* so = Ss + hl * (64 * SSTR) + j;
      int i0 = tm0 * 16 + rb;
      for (int r = 0; r < 8; ++r) {
        so[(i0 + r) * SSTR]      = acc0[r];
        so[(i0 + 16 + r) * SSTR] = acc1[r];
      }
    }
    __syncthreads();

    // Softmax rows (+ relative position bias); P f16 written in place over S.
    for (int r = tid; r < 192; r += 256) {
      int hl = r >> 6, i = r & 63;
      int hh = g * 3 + hl;
      _Float16* Prow = Ps + (hl * 64 + i) * PSTR;
      if (i >= NTOK) {
        for (int j = 0; j < 16; ++j) *(h4*)(Prow + j * 4) = (h4){};
        continue;
      }
      float* Srow = Ss + hl * (64 * SSTR) + i * SSTR;
      const int* ridx = rel_idx + i * NTOK;
      float mx = -1e30f;
      for (int j = 0; j < NTOK; ++j) {
        float s = Srow[j] + bias_table[ridx[j] * NH + hh];
        Srow[j] = s;
        mx = fmaxf(mx, s);
      }
      float sum = 0.f;
      for (int j = 0; j < NTOK; ++j) {
        float e = __expf(Srow[j] - mx);
        Srow[j] = e;
        sum += e;
      }
      float inv = 1.0f / sum;
      for (int j = 0; j < NTOK; ++j)   // read offset 4j stays ahead of write 2j
        Prow[j] = (_Float16)(Srow[j] * inv);
      for (int j = NTOK; j < 64; ++j) Prow[j] = (_Float16)0.f;
    }
    __syncthreads();

    // O = P @ V (per head 64x32, paired on M); O reuses dead xA buffer.
    for (int p = wave; p < 12; p += 8) {           // 12 pairs = 24 tiles
      int t0  = 2 * p;
      int hl  = t0 >> 3;
      int rem = t0 & 7;
      int tm0 = rem & 3, tn = rem >> 2;
      int hh  = g * 3 + hl;
      f8 acc0 = {}, acc1 = {};
      for (int kk = 0; kk < 64; kk += 32) {
        h16 vb = ldB(Vt + hh * 2560, 80, tn * 16, kk);
        h16 a0 = ldA(Ps + hl * (64 * PSTR), PSTR, tm0 * 16, kk);
        h16 a1 = ldA(Ps + hl * (64 * PSTR), PSTR, tm0 * 16 + 16, kk);
        acc0 = wmma16(a0, vb, acc0);
        acc1 = wmma16(a1, vb, acc1);
      }
      int c = hh * 32 + tn * 16 + colL;
      int tok0 = tm0 * 16 + rb, tok1 = tok0 + 16;
      for (int r = 0; r < 8; ++r) {
        xA[(tok0 + r) * XSTR + c] = (_Float16)acc0[r];
        xA[(tok1 + r) * XSTR + c] = (_Float16)acc1[r];
      }
    }
    __syncthreads();
  }

  // ---- Phase 5: y = O @ proj_w + proj_b -> compact f32 staging in Ss ----
  for (int p = wave; p < 24; p += 8) {             // 24 pairs = 48 tiles
    int tm0 = (2 * p) & 3;
    int tn  = p >> 1;                              // 0..11
    f8 acc0 = {}, acc1 = {};
    for (int kk = 0; kk < DIM; kk += 32) {
      h16 w  = ldB(pT, DIM, tn * 16, kk);
      h16 a0 = ldA(xA, XSTR, tm0 * 16, kk);
      h16 a1 = ldA(xA, XSTR, tm0 * 16 + 16, kk);
      acc0 = wmma16(a0, w, acc0);
      acc1 = wmma16(a1, w, acc1);
    }
    int col = tn * 16 + colL;
    float pb = proj_b[col];
    int tok0 = tm0 * 16 + rb, tok1 = tok0 + 16;
    float* so = Ss + col;                          // 64x192 f32, stride 192
    for (int r = 0; r < 8; ++r) {
      so[(tok0 + r) * DIM] = acc0[r] + pb;
      so[(tok1 + r) * DIM] = acc1[r] + pb;
    }
  }
  __syncthreads();

  // ---- Phase 6: coalesced float4 writeback of rows < 49 ----
  {
    const float4* src = (const float4*)Ss;
    float4* dst = (float4*)(out + (size_t)b * (NTOK * DIM));
    for (int i = tid; i < (NTOK * DIM) / 4; i += 256) dst[i] = src[i];
  }
}

extern "C" void kernel_launch(void* const* d_in, const int* in_sizes, int n_in,
                              void* d_out, int out_size, void* d_ws, size_t ws_size,
                              hipStream_t stream) {
  const float* x          = (const float*)d_in[0];
  const float* qkv_w      = (const float*)d_in[1];
  const float* qkv_b      = (const float*)d_in[2];
  const float* proj_w     = (const float*)d_in[3];
  const float* proj_b     = (const float*)d_in[4];
  const float* bias_table = (const float*)d_in[5];
  const int*   rel_idx    = (const int*)d_in[6];
  float* out = (float*)d_out;

  _Float16* wT = (_Float16*)d_ws;          // 576*192 f16 = 221184 B
  _Float16* pT = wT + 576 * 192;           // 192*192 f16 =  73728 B

  int nblk = in_sizes[0] / (NTOK * DIM);   // 4096 windows

  convert_weights<<<(576 * 192 + 255) / 256, 256, 0, stream>>>(qkv_w, proj_w, wT, pT);
  win_attn_kernel<<<nblk, 256, 0, stream>>>(x, qkv_b, proj_b, bias_table,
                                            rel_idx, wT, pT, out);
}
